// Codec_57638461112800
// MI455X (gfx1250) — matrix operand order repaired and checked by reading, not compiled, not run on hardware
//
#include <hip/hip_runtime.h>
#include <math.h>

typedef __attribute__((ext_vector_type(16))) _Float16 v16h;
typedef __attribute__((ext_vector_type(8)))  _Float16 v8h;
typedef __attribute__((ext_vector_type(8)))  float    v8f;
typedef unsigned int u32;

#define NIMG   24
#define IMG_H  512
#define IMG_W  512
#define TILE_W 64                     // central output columns per strip
#define NSTRIP (IMG_W / TILE_W)       // 8
#define COMP_W 96                     // computed columns (16-col halo each side)
#define HALO   16
#define WLDS   104                    // padded LDS row width; index i = lc+1, lc in [-1,96]
#define NWAVE  6
#define BLOCK  (NWAVE*32)             // 192
#define ROWSPLIT 4                    // vertical segments per image
#define SEG_H  (IMG_H/ROWSPLIT)       // 128
#define WARMUP 12                     // causal receptive field is 8 rows; margin
#define NBLOCKS (NSTRIP*NIMG*ROWSPLIT)  // 768

// ---- workspace layout (bytes) ----
#define OFF_WA   0
#define SZ_WA    (6*3*3*32*16*2)      // A fragments: [pred][wl][chunk][lane][16] f16 = 55296
#define OFF_BIAS (OFF_WA + SZ_WA)     // 55296
#define SZ_BIAS  (6*3*16*4)           // 1152
#define OFF_C0   (OFF_BIAS + SZ_BIAS) // 56448
#define SZ_C0    (6*144*4)            // 3456
#define SZ_WTS   (SZ_WA + SZ_BIAS + SZ_C0)  // 59904 (contiguous, copied to LDS)
#define OFF_HX   (OFF_C0 + SZ_C0)     // 59904 ; u32[24][256]
#define SZ_HX    (NIMG*256*4)
#define OFF_HD   (OFF_HX + SZ_HX)     // u32[24][2][256]
#define SZ_HD    (NIMG*2*256*4)
#define OFF_PART (OFF_HD + SZ_HD)     // float[NBLOCKS][2]
#define SZ_PART  (NBLOCKS*2*4)

// ---- dynamic LDS layout (bytes) ----
#define SM_X    0                          // float[2][WLDS]
#define SM_D1   (SM_X + 2*WLDS*4)          // float[2][WLDS][2]
#define SM_P    (SM_D1 + 2*WLDS*2*4)       // float[3][COMP_W][2] (also reduction scratch)
#define SM_HX   (SM_P + 3*COMP_W*2*4)      // u32[256]
#define SM_HD   (SM_HX + 256*4)            // u32[2][256]
#define SM_FEAT (SM_HD + 512*4)            // f16[6][3][2][WLDS][16]
#define SM_WTS  (SM_FEAT + 6*3*2*WLDS*16*2) // weights cache (A-frags | bias | c0)
#define SMEM_BYTES (SM_WTS + SZ_WTS)        // ~187.6 KB

struct Params { const float* p[72]; };   // [pred 0..5][layer c0..c3][wT,b,wL]

__device__ __forceinline__ float median3f(float a, float b, float c) {
  return fmaxf(fminf(a, fmaxf(b, c)), fminf(b, c));
}
__device__ __forceinline__ float lrelu(float v) { return v >= 0.f ? v : 0.01f*v; }
// fmod(v,2)-1 with trunc semantics (v in [-3,3] here)
__device__ __forceinline__ float wrap2(float v) { return v - 2.f*truncf(v*0.5f) - 1.f; }

__device__ __forceinline__ v8f wmma16(v16h a, v16h b, v8f c) {
  return __builtin_amdgcn_wmma_f32_16x16x32_f16(false, a, false, b, (short)0, c, false, false);
}

// ------------------------------------------------------------------
// Weight prepack: f32 params -> f16 A-fragments for wave32 WMMA
// 16-bit A 16x32 (row = lane%16, contiguous k = 16*(lane/16) + elem, as B); plus biases and c0 (VALU) params.
// K ordering: K = tap*16 + ci ; taps: 0:(-1,-1) 1:(-1,0) 2:(-1,+1) 3:(0,-1) 4:(0,0)
// ------------------------------------------------------------------
__global__ void pack_weights(Params P, _Float16* __restrict__ wA,
                             float* __restrict__ bpack, float* __restrict__ c0pack) {
  int stride = blockDim.x * gridDim.x;
  int tid0 = blockIdx.x*blockDim.x + threadIdx.x;

  for (int item = tid0; item < 6*3*3*32; item += stride) {
    int lane = item & 31;
    int rest = item >> 5;
    int chunk = rest % 3; rest /= 3;
    int wl = rest % 3; rest /= 3;       // 0..2 -> conv c1..c3 (param layer wl+1)
    int pred = rest;                     // 0..5
    const float* wT = P.p[pred*12 + (wl+1)*3 + 0];
    const float* wL = P.p[pred*12 + (wl+1)*3 + 2];
    int co_max = (wl == 2) ? 6 : 16;     // c3 has 6 output channels (zero-pad M)
    int M = lane & 15;
    bool hi = lane >= 16;
    _Float16* dst = wA + ((pred*3 + wl)*3 + chunk)*512 + lane*16;
    for (int hh = 0; hh < 16; ++hh) {
      int v = hh >> 1, pbit = hh & 1;
      int kk = (hi ? 16 : 0) + hh;
      int K = chunk*32 + kk;
      float val = 0.f;
      if (K < 80 && M < co_max) {
        int tap = K >> 4, ci = K & 15;
        val = (tap < 3) ? wT[(M*16 + ci)*3 + tap]       // wT (co,16,1,3)
                        : wL[(M*16 + ci)*2 + (tap - 3)]; // wL (co,16,1,2)
      }
      dst[hh] = (_Float16)val;
    }
  }
  for (int i = tid0; i < 6*3*16; i += stride) {
    int ch = i & 15; int rest = i >> 4;
    int wl = rest % 3; int pred = rest / 3;
    const float* b = P.p[pred*12 + (wl+1)*3 + 1];
    bpack[i] = (wl == 2 && ch >= 6) ? 0.f : b[ch];
  }
  // c0 params: per pred 144 floats: wT2[16][2][3] | wL2[16][2] | b[16]
  for (int i = tid0; i < 6*144; i += stride) {
    int pred = i / 144, o = i % 144;
    int cid = (pred < 3) ? 1 : 2;        // stage-1 ci=1, stage-2 ci=2
    const float* wT = P.p[pred*12 + 0];
    const float* b  = P.p[pred*12 + 1];
    const float* wL = P.p[pred*12 + 2];
    float val = 0.f;
    if (o < 96)      { int t = o % 3, ci = (o/3) & 1, ch = o/6; if (ci < cid) val = wT[(ch*cid + ci)*3 + t]; }
    else if (o < 128){ int oo = o - 96; int ci = oo & 1, ch = oo >> 1; if (ci < cid) val = wL[ch*cid + ci]; }
    else             { val = b[o - 128]; }
    c0pack[i] = val;
  }
}

__global__ void zero_stats(u32* __restrict__ p, int nwords) {
  int stride = blockDim.x * gridDim.x;
  for (int i = blockIdx.x*blockDim.x + threadIdx.x; i < nwords; i += stride) p[i] = 0u;
}

// ------------------------------------------------------------------
// Fused codec: both predictor stages + medians + fmod deltas + stats.
// One block per (image, 64-col strip, 128-row segment); rows streamed
// with rolling prev/curr LDS feature rows; weights cached in LDS;
// 16->16 convs on WMMA f16 (K=80 -> 3 ops). 12 warm-up rows make the
// 8-row causal receptive field exact at segment boundaries.
// ------------------------------------------------------------------
__global__ void __launch_bounds__(BLOCK)
codec_fused(const float* __restrict__ xg,
            const u32* __restrict__ wraw,            // packed weights (SZ_WTS bytes)
            u32* __restrict__ ghx, u32* __restrict__ ghd,
            float* __restrict__ part) {
  extern __shared__ char smem[];
  float* xbuf  = (float*)(smem + SM_X);
  float* d1buf = (float*)(smem + SM_D1);
  float* pbuf  = (float*)(smem + SM_P);
  u32*   lhx   = (u32*)(smem + SM_HX);
  u32*   lhd   = (u32*)(smem + SM_HD);
  _Float16* feat = (_Float16*)(smem + SM_FEAT);
  _Float16* lwA  = (_Float16*)(smem + SM_WTS);
  float* lbias   = (float*)(smem + SM_WTS + SZ_WA);
  float* lc0     = (float*)(smem + SM_WTS + SZ_WA + SZ_BIAS);

  const int tid = threadIdx.x;
  const int strip = blockIdx.x;
  const int img = blockIdx.y;
  const int seg = blockIdx.z;
  const int col0 = strip*TILE_W - HALO;               // global col of lc=0
  const float* x_img = xg + (size_t)img*IMG_H*IMG_W;

  // zero data region, then cache all weights in LDS
  for (int i = tid; i < SM_WTS/4; i += BLOCK) ((u32*)smem)[i] = 0u;
  for (int i = tid; i < SZ_WTS/4; i += BLOCK) ((u32*)(smem + SM_WTS))[i] = wraw[i];
  __syncthreads();

  const int lane = tid & 31;
  const int wid  = tid >> 5;
  const int n    = lane & 15;
  const bool hi  = lane >= 16;
  const int mbase = hi ? 8 : 0;
  const int tileBase = wid*16;

  v16h zero16;
#pragma unroll
  for (int i = 0; i < 16; ++i) zero16[i] = (_Float16)0.f;

  float ssx = 0.f, ssd = 0.f;

  const int ybase = seg*SEG_H;
  const int yend  = ybase + SEG_H;
  const int y0    = (ybase >= WARMUP) ? (ybase - WARMUP) : 0;

  for (int y = y0; y < yend; ++y) {
    const bool emit = (y >= ybase);
    const int cp = y & 1, pv = cp ^ 1;
    float* xcur  = xbuf + cp*WLDS;
    float* xprev = xbuf + pv*WLDS;
    float* d1cur  = d1buf + cp*WLDS*2;
    float* d1prev = d1buf + pv*WLDS*2;

    // ---- load x row + x statistics on central columns ----
    for (int i = tid; i < 98; i += BLOCK) {
      int lc = i - 1;
      int gc = col0 + lc;
      float v = (gc >= 0 && gc < IMG_W) ? x_img[y*IMG_W + gc] : 0.f;
      xcur[i] = v;
      if (emit && lc >= HALO && lc < HALO + TILE_W) {
        ssx += v*v;
        if (v >= -1.f && v <= 1.f) {
          int b = (int)floorf((v + 1.f)*128.f);
          b = b < 0 ? 0 : (b > 255 ? 255 : b);
          atomicAdd(&lhx[b], 1u);
        }
      }
    }
    if (tid < 4 && y + 1 < IMG_H)
      __builtin_prefetch(x_img + (size_t)(y+1)*IMG_W + col0 + HALO + tid*16, 0, 0);
    __syncthreads();

    for (int stage = 0; stage < 2; ++stage) {
      const int pb = stage*3;

      // ---- c0 (small ci -> VALU f32) ----
      for (int idx = tid; idx < 3*COMP_W*16; idx += BLOCK) {
        int ch = idx & 15;
        int lc = (idx >> 4) % COMP_W;
        int pr = (idx >> 4) / COMP_W;
        const float* W = lc0 + (pb + pr)*144;
        float acc = W[128 + ch];
        if (stage == 0) {
          acc += W[(ch*2+0)*3+0]*xprev[lc] + W[(ch*2+0)*3+1]*xprev[lc+1]
               + W[(ch*2+0)*3+2]*xprev[lc+2];
          acc += W[96 + ch*2 + 0]*xcur[lc];
        } else {
#pragma unroll
          for (int ci = 0; ci < 2; ++ci) {
            acc += W[(ch*2+ci)*3+0]*d1prev[lc*2+ci]
                 + W[(ch*2+ci)*3+1]*d1prev[(lc+1)*2+ci]
                 + W[(ch*2+ci)*3+2]*d1prev[(lc+2)*2+ci];
            acc += W[96 + ch*2 + ci]*d1cur[lc*2+ci];
          }
        }
        feat[((((pb+pr)*3 + 0)*2 + cp)*WLDS + (lc+1))*16 + ch] = (_Float16)lrelu(acc);
      }
      __syncthreads();

      // ---- c1,c2,c3 on WMMA: M=16 out-ch, N=16 cols, K=80 (3 chunks) ----
      for (int wl = 0; wl < 3; ++wl) {
        for (int pr = 0; pr < 3; ++pr) {
          const int ps = pb + pr;
          const _Float16* prevF = feat + (((ps*3 + wl)*2 + pv)*WLDS)*16;
          const _Float16* curF  = feat + (((ps*3 + wl)*2 + cp)*WLDS)*16;
          const _Float16* wbase = lwA + ((ps*3 + wl)*3)*512;
          const float* bb = lbias + (ps*3 + wl)*16;

          v8f c = *(const v8f*)(bb + mbase);   // bias-seeded accumulator

          // chunk0: taps (-1,-1)|(-1,0) ; chunk1: (-1,+1)|(0,-1) ; chunk2: (0,0)|zero
          int i0 = tileBase + n + (hi ? 0 : -1) + 1;
          v16h b0 = *(const v16h*)(prevF + i0*16);
          const _Float16* src1 = hi ? curF : prevF;
          int i1 = tileBase + n + (hi ? -1 : 1) + 1;
          v16h b1 = *(const v16h*)(src1 + i1*16);
          v16h b2 = *(const v16h*)(curF + (tileBase + n + 1)*16);
          if (hi) b2 = zero16;

          v16h a0 = *(const v16h*)(wbase + 0*512 + lane*16);
          v16h a1 = *(const v16h*)(wbase + 1*512 + lane*16);
          v16h a2 = *(const v16h*)(wbase + 2*512 + lane*16);

          c = wmma16(a0, b0, c);
          c = wmma16(a1, b1, c);
          c = wmma16(a2, b2, c);

          if (wl < 2) {
            _Float16* outF = feat + (((ps*3 + wl + 1)*2 + cp)*WLDS)*16;
            v8h pk;
#pragma unroll
            for (int r = 0; r < 8; ++r) pk[r] = (_Float16)lrelu(c[r]);
            *(v8h*)(outF + (tileBase + n + 1)*16 + mbase) = pk;
          } else if (!hi) {
            float o[6];
#pragma unroll
            for (int r = 0; r < 6; ++r) o[r] = fminf(fmaxf(c[r], -1.f), 1.f);
            int lc = tileBase + n;
            pbuf[(pr*COMP_W + lc)*2 + 0] = median3f(o[0], o[2], o[4]);
            pbuf[(pr*COMP_W + lc)*2 + 1] = median3f(o[1], o[3], o[5]);
          }
        }
        __syncthreads();
      }

      // ---- cross-predictor median + fmod delta ----
      if (stage == 0) {
        for (int idx = tid; idx < COMP_W*2; idx += BLOCK) {
          int lc = idx >> 1, ch = idx & 1;
          float p1 = median3f(pbuf[(0*COMP_W+lc)*2+ch],
                              pbuf[(1*COMP_W+lc)*2+ch],
                              pbuf[(2*COMP_W+lc)*2+ch]);
          d1cur[(lc+1)*2 + ch] = wrap2(xcur[lc+1] - p1 + 1.f);
        }
      } else {
        for (int idx = tid; idx < TILE_W*2; idx += BLOCK) {
          int lc = HALO + (idx >> 1), ch = idx & 1;
          float p2 = median3f(pbuf[(0*COMP_W+lc)*2+ch],
                              pbuf[(1*COMP_W+lc)*2+ch],
                              pbuf[(2*COMP_W+lc)*2+ch]);
          float dd = wrap2(d1cur[(lc+1)*2 + ch] - p2 + 1.f);
          if (emit) {
            ssd += dd*dd;
            if (dd >= -1.f && dd <= 1.f) {
              int b = (int)floorf((dd + 1.f)*128.f);
              b = b < 0 ? 0 : (b > 255 ? 255 : b);
              atomicAdd(&lhd[ch*256 + b], 1u);
            }
          }
        }
      }
      __syncthreads();
    } // stage
  } // y

  // ---- deterministic block reduction of squared sums (reuse pbuf) ----
  float* red = pbuf;
  red[tid] = ssx; red[256 + tid] = ssd;
  __syncthreads();
  if (tid < 64) { red[tid] += red[tid+128]; red[256+tid] += red[256+tid+128]; }
  __syncthreads();
  for (int s = 64; s > 0; s >>= 1) {
    if (tid < s) { red[tid] += red[tid+s]; red[256+tid] += red[256+tid+s]; }
    __syncthreads();
  }
  if (tid == 0) {
    float* pp = part + (((size_t)img*NSTRIP + strip)*ROWSPLIT + seg)*2;
    pp[0] = red[0]; pp[1] = red[256];
  }
  for (int i = tid; i < 256; i += BLOCK) atomicAdd(&ghx[img*256 + i], lhx[i]);
  for (int i = tid; i < 512; i += BLOCK) atomicAdd(&ghd[img*512 + i], lhd[i]);
}

// ------------------------------------------------------------------
// Finalize: entropies + RMSEs -> 4 scalars (loss1, loss0, invCR0, invCR1)
// ------------------------------------------------------------------
__global__ void finalize_kernel(const float* __restrict__ part,
                                const u32* __restrict__ hx,
                                const u32* __restrict__ hd,
                                float* __restrict__ out) {
  __shared__ float ent[72];
  int t = threadIdx.x;
  if (t < 72) {
    const u32* h = (t < 24) ? (hx + t*256) : (hd + (t-24)*256);
    const float inv_res = 1.f/(512.f*512.f);
    float e = 0.f;
    for (int i = 0; i < 256; ++i) {
      float p = (float)h[i]*inv_res;
      if (p > 0.f) e -= p*log2f(p);
    }
    ent[t] = e;
  }
  __syncthreads();
  if (t == 0) {
    float ssx = 0.f, ssd = 0.f;
    for (int i = 0; i < NBLOCKS; ++i) { ssx += part[2*i]; ssd += part[2*i+1]; }
    float ex = 0.f;
    for (int i = 0; i < 24; ++i) ex += ent[i];
    float e0 = 0.f, e1 = 0.f;
    for (int i = 0; i < 24; ++i) { e0 += ent[24 + 2*i]; e1 += ent[24 + 2*i + 1]; }
    out[0] = 128.f*sqrtf(ssd/(24.f*2.f*512.f*512.f));   // loss1
    out[1] = 128.f*sqrtf(ssx/(24.f*512.f*512.f));       // loss0 (min over 1 ch)
    out[2] = ex*(1.f/192.f);                            // invCR0
    out[3] = fminf(e0, e1)*(1.f/192.f);                 // invCR1
  }
}

extern "C" void kernel_launch(void* const* d_in, const int* in_sizes, int n_in,
                              void* d_out, int out_size, void* d_ws, size_t ws_size,
                              hipStream_t stream) {
  (void)in_sizes; (void)n_in; (void)out_size; (void)ws_size;
  char* ws = (char*)d_ws;
  Params P;
  for (int i = 0; i < 72; ++i) P.p[i] = (const float*)d_in[1 + i];

  pack_weights<<<4, 256, 0, stream>>>(P, (_Float16*)(ws + OFF_WA),
                                      (float*)(ws + OFF_BIAS), (float*)(ws + OFF_C0));

  int nwords = (SZ_HX + SZ_HD)/4;
  zero_stats<<<64, 256, 0, stream>>>((u32*)(ws + OFF_HX), nwords);

  dim3 grid(NSTRIP, NIMG, ROWSPLIT);
  codec_fused<<<grid, BLOCK, SMEM_BYTES, stream>>>(
      (const float*)d_in[0],
      (const u32*)(ws + OFF_WA),
      (u32*)(ws + OFF_HX),
      (u32*)(ws + OFF_HD),
      (float*)(ws + OFF_PART));

  finalize_kernel<<<1, 128, 0, stream>>>(
      (const float*)(ws + OFF_PART),
      (const u32*)(ws + OFF_HX),
      (const u32*)(ws + OFF_HD),
      (float*)d_out);
}
